// DecoderWithAttention_24275155157043
// MI455X (gfx1250) — matrix operand order repaired and checked by reading, not compiled, not run on hardware
//
#include <hip/hip_runtime.h>

// ---------------- types & helpers ----------------
typedef __attribute__((ext_vector_type(16))) __bf16 v16bf;
typedef __attribute__((ext_vector_type(8)))  float  v8f;

#define DEVINL __device__ __forceinline__

DEVINL __bf16 f2bf(float f) {
  unsigned u = __builtin_bit_cast(unsigned, f);
  unsigned r = (u + 0x7FFFu + ((u >> 16) & 1u)) >> 16;
  unsigned short h = (unsigned short)r;
  return __builtin_bit_cast(__bf16, h);
}

DEVINL float sigm(float x) { return 1.0f / (1.0f + __expf(-x)); }

// A-matrix 16x32 bf16 fragment (row-major source, row stride = `stride` elems)
// lane L holds row m = L&15; VGPR i pair p: k = (i>>2)*16 + (L>>4)*8 + (i&3)*2 + p
DEVINL v16bf frag_ld_a(const __bf16* __restrict__ base, int stride, int lane) {
  const __bf16* row = base + (size_t)(lane & 15) * stride + ((lane >> 4) << 3);
  v16bf f;
#pragma unroll
  for (int i = 0; i < 8; ++i) {
    int k = ((i >> 2) << 4) + ((i & 3) << 1);
    f[2 * i]     = row[k];
    f[2 * i + 1] = row[k + 1];
  }
  return f;
}

// B-matrix 32x16 bf16 fragment loaded from B^T stored [N,K] row-major.
// lane L holds col n = L&15; VGPR i pair p: k = (i>>2)*16 + (L>>4)*8 + (i&3)*2 + p
DEVINL v16bf frag_ld_b(const __bf16* __restrict__ base, int stride, int lane) {
  const __bf16* row = base + (size_t)(lane & 15) * stride + ((lane >> 4) << 3);
  v16bf f;
#pragma unroll
  for (int i = 0; i < 8; ++i) {
    int k = ((i >> 2) << 4) + ((i & 3) << 1);
    f[2 * i]     = row[k];
    f[2 * i + 1] = row[k + 1];
  }
  return f;
}

DEVINL v8f wmma_bf16(v16bf a, v16bf b, v8f c) {
  return __builtin_amdgcn_wmma_f32_16x16x32_bf16(false, a, false, b,
                                                 (short)0, c, false, false);
}

DEVINL void store_tile(float* __restrict__ C, long long ldc, int rowb, int col,
                       v8f acc, float bv) {
#pragma unroll
  for (int r = 0; r < 8; ++r)
    C[(long long)(rowb + r) * ldc + col] = acc[r] + bv;
}

// ---- NT GEMM, 64-row M-panel per wave, software-pipelined K loop ----
// C[64*Mg, N] = A[.,K] * Bt[N,K]^T + bias. Each wave: one 16-col N-tile x
// four 16-row M-tiles; B fragment reused 4x; next K-chunk's fragments are
// loaded before the WMMAs that consume the current chunk.
__global__ void gemm_nt_bf16_m64(const __bf16* __restrict__ A, int lda,
                                 const __bf16* __restrict__ Bt, int ldb,
                                 const float* __restrict__ bias,
                                 float* __restrict__ C, long long ldc,
                                 int Mgroups, int Ntiles, int K) {
  int wave = blockIdx.x * (blockDim.x >> 5) + (threadIdx.x >> 5);
  int lane = threadIdx.x & 31;
  if (wave >= Mgroups * Ntiles) return;
  int mg = wave / Ntiles, nt = wave % Ntiles;
  const __bf16* Ab = A + (size_t)mg * 64 * lda;
  const __bf16* Bb = Bt + (size_t)nt * 16 * ldb;
  const __bf16* Bpf = Bb + (size_t)(lane & 15) * ldb;
  v8f acc0 = {}, acc1 = {}, acc2 = {}, acc3 = {};

  v16bf b  = frag_ld_b(Bb, ldb, lane);
  v16bf a0 = frag_ld_a(Ab, lda, lane);
  v16bf a1 = frag_ld_a(Ab + (size_t)16 * lda, lda, lane);
  v16bf a2 = frag_ld_a(Ab + (size_t)32 * lda, lda, lane);
  v16bf a3 = frag_ld_a(Ab + (size_t)48 * lda, lda, lane);

  for (int k = 32; k < K; k += 32) {
    if (k + 224 < K) __builtin_prefetch(Bpf + k + 224, 0, 1);
    v16bf nb  = frag_ld_b(Bb + k, ldb, lane);
    v16bf na0 = frag_ld_a(Ab + k, lda, lane);
    v16bf na1 = frag_ld_a(Ab + (size_t)16 * lda + k, lda, lane);
    v16bf na2 = frag_ld_a(Ab + (size_t)32 * lda + k, lda, lane);
    v16bf na3 = frag_ld_a(Ab + (size_t)48 * lda + k, lda, lane);
    acc0 = wmma_bf16(a0, b, acc0);
    acc1 = wmma_bf16(a1, b, acc1);
    acc2 = wmma_bf16(a2, b, acc2);
    acc3 = wmma_bf16(a3, b, acc3);
    b = nb; a0 = na0; a1 = na1; a2 = na2; a3 = na3;
  }
  acc0 = wmma_bf16(a0, b, acc0);
  acc1 = wmma_bf16(a1, b, acc1);
  acc2 = wmma_bf16(a2, b, acc2);
  acc3 = wmma_bf16(a3, b, acc3);

  int col  = nt * 16 + (lane & 15);
  int rowb = mg * 64 + ((lane >> 4) << 3);
  float bv = bias ? bias[col] : 0.0f;
  store_tile(C, ldc, rowb +  0, col, acc0, bv);
  store_tile(C, ldc, rowb + 16, col, acc1, bv);
  store_tile(C, ldc, rowb + 32, col, acc2, bv);
  store_tile(C, ldc, rowb + 48, col, acc3, bv);
}

// ---- LSTM gates: G[64,2048] = X*Wih^T + H*Whh^T + b1 + b2 ----
// One wave per 16-col N-tile, full 64-row M-panel, both K loops pipelined.
// X may be split along K (concat emb | awe); split at a multiple of 32.
__global__ void lstm_gates_k(const __bf16* __restrict__ X1, int lda1,
                             const __bf16* __restrict__ X2, int lda2,
                             int K1a, int K1,
                             const __bf16* __restrict__ Wih,   // [2048, K1]
                             const __bf16* __restrict__ H, int K2,
                             const __bf16* __restrict__ Whh,   // [2048, K2]
                             const float* __restrict__ b1,
                             const float* __restrict__ b2,
                             float* __restrict__ G) {
  int wave = blockIdx.x * (blockDim.x >> 5) + (threadIdx.x >> 5);
  int lane = threadIdx.x & 31;
  if (wave >= 128) return;
  int nt = wave;
  v8f acc0 = {}, acc1 = {}, acc2 = {}, acc3 = {};

  // ---- input GEMM: X (possibly split) * Wih^T ----
  {
    const __bf16* Wb = Wih + (size_t)nt * 16 * K1;
    const __bf16* Xs = X1; int ld = lda1; int kk = 0;
    v16bf b  = frag_ld_b(Wb, K1, lane);
    v16bf a0 = frag_ld_a(Xs, ld, lane);
    v16bf a1 = frag_ld_a(Xs + (size_t)16 * ld, ld, lane);
    v16bf a2 = frag_ld_a(Xs + (size_t)32 * ld, ld, lane);
    v16bf a3 = frag_ld_a(Xs + (size_t)48 * ld, ld, lane);
    for (int k = 32; k < K1; k += 32) {
      if (X2 != nullptr && k >= K1a) { Xs = X2; ld = lda2; kk = k - K1a; }
      else                           { Xs = X1; ld = lda1; kk = k; }
      v16bf nb  = frag_ld_b(Wb + k, K1, lane);
      v16bf na0 = frag_ld_a(Xs + kk, ld, lane);
      v16bf na1 = frag_ld_a(Xs + (size_t)16 * ld + kk, ld, lane);
      v16bf na2 = frag_ld_a(Xs + (size_t)32 * ld + kk, ld, lane);
      v16bf na3 = frag_ld_a(Xs + (size_t)48 * ld + kk, ld, lane);
      acc0 = wmma_bf16(a0, b, acc0);
      acc1 = wmma_bf16(a1, b, acc1);
      acc2 = wmma_bf16(a2, b, acc2);
      acc3 = wmma_bf16(a3, b, acc3);
      b = nb; a0 = na0; a1 = na1; a2 = na2; a3 = na3;
    }
    acc0 = wmma_bf16(a0, b, acc0);
    acc1 = wmma_bf16(a1, b, acc1);
    acc2 = wmma_bf16(a2, b, acc2);
    acc3 = wmma_bf16(a3, b, acc3);
  }

  // ---- recurrent GEMM: H * Whh^T ----
  {
    const __bf16* W2b = Whh + (size_t)nt * 16 * K2;
    v16bf b  = frag_ld_b(W2b, K2, lane);
    v16bf a0 = frag_ld_a(H, K2, lane);
    v16bf a1 = frag_ld_a(H + (size_t)16 * K2, K2, lane);
    v16bf a2 = frag_ld_a(H + (size_t)32 * K2, K2, lane);
    v16bf a3 = frag_ld_a(H + (size_t)48 * K2, K2, lane);
    for (int k = 32; k < K2; k += 32) {
      v16bf nb  = frag_ld_b(W2b + k, K2, lane);
      v16bf na0 = frag_ld_a(H + k, K2, lane);
      v16bf na1 = frag_ld_a(H + (size_t)16 * K2 + k, K2, lane);
      v16bf na2 = frag_ld_a(H + (size_t)32 * K2 + k, K2, lane);
      v16bf na3 = frag_ld_a(H + (size_t)48 * K2 + k, K2, lane);
      acc0 = wmma_bf16(a0, b, acc0);
      acc1 = wmma_bf16(a1, b, acc1);
      acc2 = wmma_bf16(a2, b, acc2);
      acc3 = wmma_bf16(a3, b, acc3);
      b = nb; a0 = na0; a1 = na1; a2 = na2; a3 = na3;
    }
    acc0 = wmma_bf16(a0, b, acc0);
    acc1 = wmma_bf16(a1, b, acc1);
    acc2 = wmma_bf16(a2, b, acc2);
    acc3 = wmma_bf16(a3, b, acc3);
  }

  int col  = nt * 16 + (lane & 15);
  int rowb = (lane >> 4) << 3;
  float bv = b1[col] + b2[col];
  store_tile(G, 2048, rowb +  0, col, acc0, bv);
  store_tile(G, 2048, rowb + 16, col, acc1, bv);
  store_tile(G, 2048, rowb + 32, col, acc2, bv);
  store_tile(G, 2048, rowb + 48, col, acc3, bv);
}

// ---------------- elementwise LSTM cell (gate order i,f,g,o) ----------------
__global__ void lstm_cell_k(const float* __restrict__ G,
                            float* __restrict__ c,
                            __bf16* __restrict__ hb) {
  int idx = blockIdx.x * blockDim.x + threadIdx.x;
  if (idx >= 64 * 512) return;
  int b = idx >> 9, j = idx & 511;
  const float* g = G + (size_t)b * 2048;
  float gi = sigm(g[j]);
  float gf = sigm(g[512 + j]);
  float gg = tanhf(g[1024 + j]);
  float go = sigm(g[1536 + j]);
  float cn = gf * c[idx] + gi * gg;
  c[idx]  = cn;
  hb[idx] = f2bf(go * tanhf(cn));
}

// ---------------- attention scores: e[b,p] = sum_a relu(att1+att2)*Wfa + bfa ----------
__global__ void attn_e_k(const float* __restrict__ att1,
                         const float* __restrict__ att2,
                         const float* __restrict__ Wfa,
                         const float* __restrict__ bfa,
                         float* __restrict__ e) {
  int wave = blockIdx.x * (blockDim.x >> 5) + (threadIdx.x >> 5);
  int lane = threadIdx.x & 31;
  if (wave >= 64 * 196) return;
  int b = wave / 196, p = wave % 196;
  const float* a1 = att1 + ((size_t)b * 196 + p) * 512;
  const float* a2 = att2 + (size_t)b * 512;
  float s = 0.0f;
  for (int a = lane; a < 512; a += 32) {
    float v = a1[a] + a2[a];
    s += (v > 0.0f ? v : 0.0f) * Wfa[a];
  }
#pragma unroll
  for (int off = 16; off; off >>= 1) s += __shfl_xor(s, off, 32);
  if (lane == 0) e[wave] = s + bfa[0];
}

// ---------------- softmax over P=196 + context awe (bf16 out) ----------------
__global__ void softmax_awe_k(const float* __restrict__ e,
                              const float* __restrict__ enc,
                              __bf16* __restrict__ aweb) {
  __shared__ float alpha[196];
  __shared__ float red[256];
  int b = blockIdx.x, tid = threadIdx.x;
  const float* eb = e + (size_t)b * 196;
  float m = -1e30f;
  for (int p = tid; p < 196; p += 256) m = fmaxf(m, eb[p]);
  red[tid] = m; __syncthreads();
  for (int s = 128; s; s >>= 1) { if (tid < s) red[tid] = fmaxf(red[tid], red[tid + s]); __syncthreads(); }
  m = red[0]; __syncthreads();
  float sum = 0.0f;
  for (int p = tid; p < 196; p += 256) { float v = __expf(eb[p] - m); alpha[p] = v; sum += v; }
  red[tid] = sum; __syncthreads();
  for (int s = 128; s; s >>= 1) { if (tid < s) red[tid] += red[tid + s]; __syncthreads(); }
  float inv = 1.0f / red[0];
  __syncthreads();
  const float* encb = enc + (size_t)b * 196 * 512;
  for (int d = tid; d < 512; d += 256) {
    float s2 = 0.0f;
    for (int p = 0; p < 196; ++p) s2 += alpha[p] * encb[(size_t)p * 512 + d];
    aweb[(size_t)b * 512 + d] = f2bf(s2 * inv);
  }
}

// ---------------- setup kernels ----------------
__global__ void conv_bf16_k(const float* __restrict__ s, __bf16* __restrict__ d, int n) {
  int i = blockIdx.x * blockDim.x + threadIdx.x;
  if (i < n) d[i] = f2bf(s[i]);
}

// s [K,N] fp32 -> d [N,K] bf16
__global__ void tconv_bf16_k(const float* __restrict__ s, __bf16* __restrict__ d,
                             int K, int N) {
  int i = blockIdx.x * blockDim.x + threadIdx.x;
  if (i >= K * N) return;
  int n = i / K, k = i % K;
  d[i] = f2bf(s[(size_t)k * N + n]);
}

__global__ void mean_enc_k(const float* __restrict__ enc, float* __restrict__ me) {
  int idx = blockIdx.x * blockDim.x + threadIdx.x;
  if (idx >= 64 * 512) return;
  int b = idx >> 9, d = idx & 511;
  const float* p = enc + (size_t)b * 196 * 512 + d;
  float s = 0.0f;
  for (int i = 0; i < 196; ++i) s += p[(size_t)i * 512];
  me[idx] = s * (1.0f / 196.0f);
}

__global__ void init_state_k(const float* __restrict__ me,
                             const float* __restrict__ Wih, const float* __restrict__ bih,
                             const float* __restrict__ Wic, const float* __restrict__ bic,
                             __bf16* __restrict__ h0b, __bf16* __restrict__ h1b,
                             float* __restrict__ c0, float* __restrict__ c1) {
  int idx = blockIdx.x * blockDim.x + threadIdx.x;
  if (idx >= 64 * 512) return;
  int b = idx >> 9, j = idx & 511;
  const float* m = me + (size_t)b * 512;
  float sh = 0.0f, sc = 0.0f;
  for (int k = 0; k < 512; ++k) {
    float mv = m[k];
    sh += mv * Wih[(size_t)k * 512 + j];
    sc += mv * Wic[(size_t)k * 512 + j];
  }
  sh += bih[j]; sc += bic[j];
  __bf16 hb = f2bf(sh);
  h0b[idx] = hb; h1b[idx] = hb;
  c0[idx] = sc;  c1[idx] = sc;
}

// xemb[t][b][j] = bf16(emb[captions[b][t]][j])
__global__ void gather_emb_k(const float* __restrict__ emb,
                             const int* __restrict__ caps,
                             __bf16* __restrict__ xemb) {
  int idx = blockIdx.x * blockDim.x + threadIdx.x;
  if (idx >= 20 * 64 * 512) return;
  int j = idx & 511;
  int bt = idx >> 9;
  int t = bt / 64, b = bt % 64;
  int tok = caps[b * 20 + t];
  xemb[idx] = f2bf(emb[(size_t)tok * 512 + j]);
}

// ---------------- host launcher ----------------
extern "C" void kernel_launch(void* const* d_in, const int* in_sizes, int n_in,
                              void* d_out, int out_size, void* d_ws, size_t ws_size,
                              hipStream_t stream) {
  const float* enc  = (const float*)d_in[0];   // [64,196,512]
  const int*   caps = (const int*)d_in[1];     // [64,20]
  const float* emb  = (const float*)d_in[3];   // [32000,512]
  const float* Wea  = (const float*)d_in[4];
  const float* bea  = (const float*)d_in[5];
  const float* Wda  = (const float*)d_in[6];
  const float* bda  = (const float*)d_in[7];
  const float* Wfa  = (const float*)d_in[8];
  const float* bfa  = (const float*)d_in[9];
  const float* Wih0 = (const float*)d_in[10];  // [2048,1024]
  const float* Whh0 = (const float*)d_in[11];  // [2048,512]
  const float* bih0 = (const float*)d_in[12];
  const float* bhh0 = (const float*)d_in[13];
  const float* Wih1 = (const float*)d_in[14];  // [2048,512]
  const float* Whh1 = (const float*)d_in[15];  // [2048,512]
  const float* bih1 = (const float*)d_in[16];
  const float* bhh1 = (const float*)d_in[17];
  const float* Wfc  = (const float*)d_in[18];  // [512,32000]
  const float* bfc  = (const float*)d_in[19];
  const float* Wih  = (const float*)d_in[20];
  const float* bih  = (const float*)d_in[21];
  const float* Wic  = (const float*)d_in[22];
  const float* bic  = (const float*)d_in[23];
  float* out = (float*)d_out;                  // [64,20,32000]

  char* p = (char*)d_ws;
  auto alloc = [&](size_t bytes) -> void* {
    void* r = (void*)p;
    p += (bytes + 255) & ~(size_t)255;
    return r;
  };

  __bf16* WfcT  = (__bf16*)alloc(32000ull * 512 * 2); // [32000,512]
  __bf16* WdaT  = (__bf16*)alloc(512ull * 512 * 2);   // [512,512]
  __bf16* WeaT  = (__bf16*)alloc(512ull * 512 * 2);
  __bf16* Wih0b = (__bf16*)alloc(2048ull * 1024 * 2);
  __bf16* Whh0b = (__bf16*)alloc(2048ull * 512 * 2);
  __bf16* Wih1b = (__bf16*)alloc(2048ull * 512 * 2);
  __bf16* Whh1b = (__bf16*)alloc(2048ull * 512 * 2);
  __bf16* encb  = (__bf16*)alloc(64ull * 196 * 512 * 2);
  float*  att1  = (float*)alloc(64ull * 196 * 512 * 4);
  __bf16* xemb  = (__bf16*)alloc(20ull * 64 * 512 * 2);
  float*  att2  = (float*)alloc(64ull * 512 * 4);
  float*  evec  = (float*)alloc(64ull * 196 * 4);
  __bf16* aweb  = (__bf16*)alloc(64ull * 512 * 2);
  float*  G0    = (float*)alloc(64ull * 2048 * 4);
  float*  G1    = (float*)alloc(64ull * 2048 * 4);
  float*  c0f   = (float*)alloc(64ull * 512 * 4);
  float*  c1f   = (float*)alloc(64ull * 512 * 4);
  __bf16* h0b   = (__bf16*)alloc(64ull * 512 * 2);
  __bf16* h1b   = (__bf16*)alloc(64ull * 512 * 2);
  float*  menc  = (float*)alloc(64ull * 512 * 4);
  (void)ws_size; (void)in_sizes; (void)n_in; (void)out_size;

  const int TB = 256;
  auto grid1 = [](int n) { return (n + 255) / 256; };
  auto gridw = [](int waves) { return (waves + 7) / 8; }; // 8 waves / 256-thr block

  // ---- setup ----
  tconv_bf16_k<<<grid1(512 * 32000), TB, 0, stream>>>(Wfc, WfcT, 512, 32000);
  tconv_bf16_k<<<grid1(512 * 512),   TB, 0, stream>>>(Wda, WdaT, 512, 512);
  tconv_bf16_k<<<grid1(512 * 512),   TB, 0, stream>>>(Wea, WeaT, 512, 512);
  conv_bf16_k<<<grid1(2048 * 1024),  TB, 0, stream>>>(Wih0, Wih0b, 2048 * 1024);
  conv_bf16_k<<<grid1(2048 * 512),   TB, 0, stream>>>(Whh0, Whh0b, 2048 * 512);
  conv_bf16_k<<<grid1(2048 * 512),   TB, 0, stream>>>(Wih1, Wih1b, 2048 * 512);
  conv_bf16_k<<<grid1(2048 * 512),   TB, 0, stream>>>(Whh1, Whh1b, 2048 * 512);
  conv_bf16_k<<<grid1(64 * 196 * 512), TB, 0, stream>>>(enc, encb, 64 * 196 * 512);
  gather_emb_k<<<grid1(20 * 64 * 512), TB, 0, stream>>>(emb, caps, xemb);
  mean_enc_k<<<grid1(64 * 512), TB, 0, stream>>>(enc, menc);
  init_state_k<<<grid1(64 * 512), TB, 0, stream>>>(menc, Wih, bih, Wic, bic,
                                                   h0b, h1b, c0f, c1f);
  // att1 = enc @ Wea + bea : M=12544 (196 groups of 64), N=512 (32 tiles), K=512
  gemm_nt_bf16_m64<<<gridw(196 * 32), TB, 0, stream>>>(encb, 512, WeaT, 512, bea,
                                                       att1, 512, 196, 32, 512);

  // ---- timestep loop ----
  for (int t = 0; t < 20; ++t) {
    // att2 = h1 @ Wda + bda  (1 group x 32 N-tiles, K=512)
    gemm_nt_bf16_m64<<<gridw(32), TB, 0, stream>>>(h1b, 512, WdaT, 512, bda,
                                                   att2, 512, 1, 32, 512);
    // e[b,p]
    attn_e_k<<<gridw(64 * 196), TB, 0, stream>>>(att1, att2, Wfa, bfa, evec);
    // softmax + context
    softmax_awe_k<<<64, TB, 0, stream>>>(evec, enc, aweb);
    // LSTM layer 0 gates: X = [emb_t | awe], K1=1024 split at 512
    const __bf16* xe = xemb + (size_t)t * 64 * 512;
    lstm_gates_k<<<gridw(128), TB, 0, stream>>>(xe, 512, aweb, 512, 512, 1024,
                                                Wih0b, h0b, 512, Whh0b,
                                                bih0, bhh0, G0);
    lstm_cell_k<<<grid1(64 * 512), TB, 0, stream>>>(G0, c0f, h0b);
    // LSTM layer 1 gates
    lstm_gates_k<<<gridw(128), TB, 0, stream>>>(h0b, 512, nullptr, 0, 512, 512,
                                                Wih1b, h1b, 512, Whh1b,
                                                bih1, bhh1, G1);
    lstm_cell_k<<<grid1(64 * 512), TB, 0, stream>>>(G1, c1f, h1b);
    // pred = h1 @ Wfc + bfc -> out[b, t, :]   (1 group x 2000 N-tiles, K=512)
    gemm_nt_bf16_m64<<<gridw(2000), TB, 0, stream>>>(h1b, 512, WfcT, 512, bfc,
                                                     out + (size_t)t * 32000,
                                                     (long long)20 * 32000,
                                                     1, 2000, 512);
  }
}
